// O3TensorProductFC_78030965833798
// MI455X (gfx1250) — compile-verified, matrix operand order checked by reading, not run on hardware
//
#include <hip/hip_runtime.h>

typedef __attribute__((ext_vector_type(16))) _Float16     v16h;
typedef __attribute__((ext_vector_type(8)))  float        v8f;
typedef __attribute__((ext_vector_type(4)))  unsigned int u32x4;
typedef __attribute__((ext_vector_type(8)))  int          i32x8;
typedef __attribute__((ext_vector_type(4)))  int          i32x4;

#define CCH 128
#define INV_SQRT3 0.5773502691896258f

// Padded LDS layout for staged weights: each 128-half (256B) row gets 16B of
// padding -> row stride 136 halfs (272B = 68 DWORDs) so lane-indexed column
// reads spread across LDS banks instead of hitting one bank group.
#define LDS_ROW   136
#define LDS_MAT   (128 * LDS_ROW)        // halfs per 128x128 matrix
#define LDS_BYTES (4 * LDS_MAT * 2)      // 139264 B: 4 matrices + padding
#define NBLOCKS   2048                   // 8 tiles per block at N = 262144

// ---------------------------------------------------------------------------
// Prep: transpose 128x128 fp32 weights -> f16 (Wt[w][c]), folding 1/sqrt(3)
// into W_vv. d_ws layout (f16): [0) Wss^T  [16384) Wvv^T  [32768) Wsv^T
// [49152) Wvs^T  -- one contiguous 128 KB block, TDM-staged into LDS later.
// ---------------------------------------------------------------------------
__global__ void o3tp_wconv(const float* __restrict__ wss, const float* __restrict__ wsv,
                           const float* __restrict__ wvs, const float* __restrict__ wvv,
                           _Float16* __restrict__ wt) {
  int idx = blockIdx.x * blockDim.x + threadIdx.x;  // 0..16383
  int w = idx >> 7;
  int c = idx & 127;
  int src = c * CCH + w;
  wt[idx]         = (_Float16)wss[src];
  wt[16384 + idx] = (_Float16)(wvv[src] * INV_SQRT3);
  wt[32768 + idx] = (_Float16)wsv[src];
  wt[49152 + idx] = (_Float16)wvs[src];
}

// B fragment (32x16 K-major f16) from padded LDS: lane holds column
// n = lane&15; e<8 -> K = kb*32 + 8*half + e ; e>=8 -> +16.
__device__ __forceinline__ v16h o3tp_load_b(const _Float16* __restrict__ m,
                                            int n, int kb, int half) {
  const _Float16* p = m + n * LDS_ROW + kb * 32 + half * 8;
  v16h b;
#pragma unroll
  for (int e = 0; e < 8; ++e)  b[e] = p[e];
#pragma unroll
  for (int e = 8; e < 16; ++e) b[e] = p[e + 8];
  return b;
}

// ---------------------------------------------------------------------------
// Main kernel: block = 4 waves, loops over tiles (16 samples each).
//   wave 0  : out_s   = (xs*ys)@Wss + (xv.yv)@(Wvv/sqrt3) + b
//   wave 1+i: out_v_i = (yv_i*xs)@Wsv + (ys*xv_i)@Wvs, i = 0..2
// All B fragments come from LDS (TDM-staged, padded rows).
// ---------------------------------------------------------------------------
__global__ __launch_bounds__(128) void o3tp_main(const float* __restrict__ x,
                                                 const float* __restrict__ y,
                                                 const float* __restrict__ bias,
                                                 const _Float16* __restrict__ wt,
                                                 float* __restrict__ out,
                                                 int ntiles) {
  extern __shared__ _Float16 smem[];
  const int tid  = threadIdx.x;
  const int lane = tid & 31;
  const int wave = tid >> 5;
  const int nl   = lane & 15;
  const int half = lane >> 4;

  // ---- stage all 4 transposed f16 weight matrices (128 KB) into padded LDS
#if defined(__AMDGCN__) && __has_builtin(__builtin_amdgcn_tensor_load_to_lds)
  if (wave == 0) {
    // D# group 0: count=1, lds_addr, 57b global addr, type=2 ("image")
    u32x4 g0;
    unsigned long long ga = (unsigned long long)(uintptr_t)wt;
    g0[0] = 1u;
    g0[1] = (unsigned)(uintptr_t)smem;       // LDS byte offset (addr[31:0])
    g0[2] = (unsigned)ga;
    g0[3] = (unsigned)((ga >> 32) & 0x1FFFFFFull) | (2u << 30);
    // D# group 1: 8B elements, 1-D 16384-elem tile, pad 4 DWORDs per 64 DWORDs
    // (=> 16B pad after every 256B row => LDS_ROW stride), no multicast.
    i32x8 g1 = (i32x8)0;
    g1[0] = (3 << 16) | (1 << 20) | (5 << 22) | (3 << 25);
    g1[1] = (int)((16384u & 0xFFFFu) << 16); // tensor_dim0 = 16384 (low 16)
    g1[2] = 0;                               // tensor_dim0 hi / tensor_dim1 lo
    g1[3] = (int)(16384u << 16);             // tile_dim0 = 16384
    g1[4] = 0;                               // tile_dim1/2 unused
    g1[5] = 16384;                           // tensor_dim0_stride
    i32x4 z4 = (i32x4)0;
#if __clang_major__ >= 23
    i32x8 z8 = (i32x8)0;
    __builtin_amdgcn_tensor_load_to_lds(g0, g1, z4, z4, z8, 0);
#else
    __builtin_amdgcn_tensor_load_to_lds(g0, g1, z4, z4, 0);
#endif
    __builtin_amdgcn_s_wait_tensorcnt(0);
  }
#else
  {   // fallback: cooperative copy with identical padded layout
    const float4* src = (const float4*)wt;
    for (int c = tid; c < 8192; c += 128) {   // 8192 x 16B = 128 KB
      int row = c >> 4;
      int col = c & 15;
      ((float4*)smem)[row * 17 + col] = src[c];
    }
  }
#endif
  __syncthreads();

  const _Float16* wss_s = smem;
  const _Float16* wvv_s = smem + LDS_MAT;
  const _Float16* wsv_s = smem + 2 * LDS_MAT;
  const _Float16* wvs_s = smem + 3 * LDS_MAT;

  // bias depends only on channel: preload once per lane (wave 0 only)
  float bv[8];
  if (wave == 0) {
#pragma unroll
    for (int jn = 0; jn < 8; ++jn) bv[jn] = bias[jn * 16 + nl];
  }
  const int i = wave - 1;  // vector component for waves 1..3

  for (int tile = blockIdx.x; tile < ntiles; tile += gridDim.x) {
    const int sample = tile * 16 + nl;
    const float* __restrict__ xrow = x + (long)sample * (4 * CCH);
    const float4 y4 = *(const float4*)(y + (long)sample * 4);
    const float ys  = y4.x;

    v8f acc[8];

    if (wave == 0) {
      // ---- out_s ----
#pragma unroll
      for (int jn = 0; jn < 8; ++jn)
#pragma unroll
        for (int r = 0; r < 8; ++r) acc[jn][r] = bv[jn];

#pragma unroll
      for (int kb = 0; kb < 4; ++kb) {
        const int kbase = kb * 32 + half * 8;
        v16h a_ss, a_vv;
#pragma unroll
        for (int e = 0; e < 16; ++e) {
          int k = kbase + e + (e >= 8 ? 8 : 0);
          float xs  = xrow[k];
          float xv0 = xrow[CCH + 3 * k + 0];
          float xv1 = xrow[CCH + 3 * k + 1];
          float xv2 = xrow[CCH + 3 * k + 2];
          a_ss[e] = (_Float16)(xs * ys);
          a_vv[e] = (_Float16)(xv0 * y4.y + xv1 * y4.z + xv2 * y4.w);
        }
#pragma unroll
        for (int jn = 0; jn < 8; ++jn) {
          v16h b1 = o3tp_load_b(wss_s, jn * 16 + nl, kb, half);
          acc[jn] = __builtin_amdgcn_wmma_f32_16x16x32_f16(
              false, a_ss, false, b1, (short)0, acc[jn], false, false);
          v16h b2 = o3tp_load_b(wvv_s, jn * 16 + nl, kb, half);
          acc[jn] = __builtin_amdgcn_wmma_f32_16x16x32_f16(
              false, a_vv, false, b2, (short)0, acc[jn], false, false);
        }
      }
#pragma unroll
      for (int jn = 0; jn < 8; ++jn) {
#pragma unroll
        for (int r = 0; r < 8; ++r) {
          int m = r + half * 8;
          out[(long)(tile * 16 + m) * (4 * CCH) + jn * 16 + nl] = acc[jn][r];
        }
      }
    } else {
      // ---- out_v_i ----
      const float yvi = (i == 0) ? y4.y : (i == 1 ? y4.z : y4.w);
#pragma unroll
      for (int jn = 0; jn < 8; ++jn)
#pragma unroll
        for (int r = 0; r < 8; ++r) acc[jn][r] = 0.0f;

#pragma unroll
      for (int kb = 0; kb < 4; ++kb) {
        const int kbase = kb * 32 + half * 8;
        v16h a_sv, a_vs;
#pragma unroll
        for (int e = 0; e < 16; ++e) {
          int k = kbase + e + (e >= 8 ? 8 : 0);
          a_sv[e] = (_Float16)(xrow[k] * yvi);
          a_vs[e] = (_Float16)(xrow[CCH + 3 * k + i] * ys);
        }
#pragma unroll
        for (int jn = 0; jn < 8; ++jn) {
          v16h b1 = o3tp_load_b(wsv_s, jn * 16 + nl, kb, half);
          acc[jn] = __builtin_amdgcn_wmma_f32_16x16x32_f16(
              false, a_sv, false, b1, (short)0, acc[jn], false, false);
          v16h b2 = o3tp_load_b(wvs_s, jn * 16 + nl, kb, half);
          acc[jn] = __builtin_amdgcn_wmma_f32_16x16x32_f16(
              false, a_vs, false, b2, (short)0, acc[jn], false, false);
        }
      }
#pragma unroll
      for (int jn = 0; jn < 8; ++jn) {
#pragma unroll
        for (int r = 0; r < 8; ++r) {
          int m = r + half * 8;
          int w = jn * 16 + nl;
          out[(long)(tile * 16 + m) * (4 * CCH) + CCH + w * 3 + i] = acc[jn][r];
        }
      }
    }
  }
}

extern "C" void kernel_launch(void* const* d_in, const int* in_sizes, int n_in,
                              void* d_out, int out_size, void* d_ws, size_t ws_size,
                              hipStream_t stream) {
  const float* x    = (const float*)d_in[0];
  const float* y    = (const float*)d_in[1];
  const float* w_ss = (const float*)d_in[2];
  const float* w_sv = (const float*)d_in[3];
  const float* w_vs = (const float*)d_in[4];
  const float* w_vv = (const float*)d_in[5];
  const float* b    = (const float*)d_in[6];
  _Float16* wt = (_Float16*)d_ws;           // needs 128 KB of workspace
  float* out = (float*)d_out;
  (void)ws_size; (void)n_in; (void)out_size;

  const int n_samples = in_sizes[0] / (4 * CCH);
  const int n_tiles   = n_samples / 16;
  const int grid      = n_tiles < NBLOCKS ? n_tiles : NBLOCKS;

  // allow >64KB dynamic LDS (gfx1250 WGP: up to 320KB per workgroup)
  (void)hipFuncSetAttribute((const void*)o3tp_main,
                            hipFuncAttributeMaxDynamicSharedMemorySize, LDS_BYTES);

  o3tp_wconv<<<(CCH * CCH) / 256, 256, 0, stream>>>(w_ss, w_sv, w_vs, w_vv, wt);
  o3tp_main<<<grid, 128, LDS_BYTES, stream>>>(x, y, b, wt, out, n_tiles);
}
